// TransformerBlock_59296318488709
// MI455X (gfx1250) — matrix operand order repaired and checked by reading, not compiled, not run on hardware
//
#include <hip/hip_runtime.h>
#include <hip/hip_bf16.h>
#include <math.h>

#define EMB   1024
#define HEADS 16
#define HDIM  64
#define DFF   4096
#define SEQ   2048
#define BATCH 2
#define NTOK  (BATCH * SEQ)   // 4096 tokens

typedef __attribute__((ext_vector_type(16))) __bf16 v16bf;
typedef __attribute__((ext_vector_type(8)))  __bf16 v8bf;
typedef __attribute__((ext_vector_type(8)))  float  v8f;

// ---------------------------------------------------------------------------
// WMMA helper: D = A(16x32 bf16) * B(32x16 bf16) + C (f32)
// ---------------------------------------------------------------------------
__device__ inline v8f wmma_bf16(v16bf a, v16bf b, v8f c) {
  return __builtin_amdgcn_wmma_f32_16x16x32_bf16(
      /*neg_a=*/false, a, /*neg_b=*/false, b,
      /*c_mod=*/(short)0, c, /*reuse_a=*/false, /*reuse_b=*/false);
}

__device__ inline float gelu_tanh_f(float x) {
  float x3 = x * x * x;
  return 0.5f * x * (1.0f + tanhf(0.7978845608028654f * (x + 0.044715f * x3)));
}

// ---------------------------------------------------------------------------
// Weight convert + transpose: dst[c*R + r] = (bf16) src[r*C + c]
// ---------------------------------------------------------------------------
__global__ void k_cvt_transpose(const float* __restrict__ src,
                                __bf16* __restrict__ dst, int R, int C) {
  size_t idx = (size_t)blockIdx.x * 256 + threadIdx.x;
  size_t total = (size_t)R * C;
  if (idx >= total) return;
  int r = (int)(idx / C);
  int c = (int)(idx % C);
  dst[(size_t)c * R + r] = (__bf16)src[idx];
}

// ---------------------------------------------------------------------------
// LayerNorm over E=1024, one block (256 thr) per token, output bf16
// ---------------------------------------------------------------------------
__global__ void k_layernorm(const float* __restrict__ x,
                            const float* __restrict__ g,
                            const float* __restrict__ sh,
                            __bf16* __restrict__ out) {
  const int t = blockIdx.x;
  const float* row = x + (size_t)t * EMB;
  const int tid = threadIdx.x;
  __shared__ float red[8];
  float v[4];
  float ps = 0.f;
#pragma unroll
  for (int i = 0; i < 4; i++) { v[i] = row[tid + 256 * i]; ps += v[i]; }
#pragma unroll
  for (int off = 16; off >= 1; off >>= 1) ps += __shfl_xor(ps, off, 32);
  if ((tid & 31) == 0) red[tid >> 5] = ps;
  __syncthreads();
  float mean = 0.f;
#pragma unroll
  for (int i = 0; i < 8; i++) mean += red[i];
  mean *= (1.0f / EMB);
  __syncthreads();
  float pv = 0.f;
#pragma unroll
  for (int i = 0; i < 4; i++) { float d = v[i] - mean; pv += d * d; }
#pragma unroll
  for (int off = 16; off >= 1; off >>= 1) pv += __shfl_xor(pv, off, 32);
  if ((tid & 31) == 0) red[tid >> 5] = pv;
  __syncthreads();
  float var = 0.f;
#pragma unroll
  for (int i = 0; i < 8; i++) var += red[i];
  var *= (1.0f / EMB);
  float rstd = rsqrtf(var + 1e-5f);
#pragma unroll
  for (int i = 0; i < 4; i++) {
    int c = tid + 256 * i;
    out[(size_t)t * EMB + c] = (__bf16)(g[c] * (v[i] - mean) * rstd + sh[c]);
  }
}

// ---------------------------------------------------------------------------
// bf16 GEMM: C[M,N] = A[M,K] (row-major bf16) * B (stored transposed Bt[N,K])
// 8 waves/block (2 M x 4 N-strips). Each wave: 16(M) x 64(N) strip = 4 accs,
// A fragment loaded once per K-step and reused across 4 WMMAs.
// Block tile: 32(M) x 256(N).
// MODE 0: bf16 row-major out (+bias, opt GELU)
// MODE 1: bf16 out transposed per head -> Vt[B][H][64][SEQ]  (N==EMB, M==NTOK)
// MODE 2: f32 out = resid + acc + bias
// ---------------------------------------------------------------------------
template <int MODE, bool GELU>
__global__ __launch_bounds__(256) void k_gemm(const __bf16* __restrict__ A,
                                              const __bf16* __restrict__ Bt,
                                              const float* __restrict__ bias,
                                              const float* __restrict__ resid,
                                              void* __restrict__ out, int M,
                                              int N, int K) {
  const int wave = threadIdx.x >> 5;
  const int lane = threadIdx.x & 31;
  const int hl   = lane >> 4;      // half-wave: 0 or 1
  const int l16  = lane & 15;
  const int m0 = blockIdx.y * 32 + (wave >> 2) * 16;
  const int n0 = blockIdx.x * 256 + (wave & 3) * 64;

  const __bf16* arow = A  + (size_t)(m0 + l16) * K + hl * 16;       // A frag
  const __bf16* brow = Bt + (size_t)(n0 + l16) * K + hl * 16;       // B strip
  const size_t bstep = (size_t)16 * K;                              // 16 rows

  v8f acc0 = {}, acc1 = {}, acc2 = {}, acc3 = {};
  for (int kb = 0; kb < K; kb += 32) {
    v8bf a0 = *(const v8bf*)(arow + kb);        // K = base + 16h + 0..7
    v8bf a1 = *(const v8bf*)(arow + kb + 8);    // K = base + 16h + 8..15
    v16bf a = __builtin_shufflevector(a0, a1, 0, 1, 2, 3, 4, 5, 6, 7,
                                      8, 9, 10, 11, 12, 13, 14, 15);
    v16bf b0 = *(const v16bf*)(brow + kb);
    v16bf b1 = *(const v16bf*)(brow + bstep + kb);
    v16bf b2 = *(const v16bf*)(brow + 2 * bstep + kb);
    v16bf b3 = *(const v16bf*)(brow + 3 * bstep + kb);
    acc0 = wmma_bf16(a, b0, acc0);
    acc1 = wmma_bf16(a, b1, acc1);
    acc2 = wmma_bf16(a, b2, acc2);
    acc3 = wmma_bf16(a, b3, acc3);
  }

  v8f accs[4] = {acc0, acc1, acc2, acc3};
#pragma unroll
  for (int t = 0; t < 4; t++) {
    const int n = n0 + t * 16 + l16;
    const float bv = bias ? bias[n] : 0.f;
#pragma unroll
    for (int r = 0; r < 8; r++) {
      const int m = m0 + r + hl * 8;
      float v = accs[t][r] + bv;
      if (GELU) v = gelu_tanh_f(v);
      if (MODE == 0) {
        ((__bf16*)out)[(size_t)m * N + n] = (__bf16)v;
      } else if (MODE == 1) {
        const int b_ = m >> 11;           // token / SEQ  (SEQ = 2048)
        const int s_ = m & (SEQ - 1);
        const int hh = n >> 6;            // col / 64
        const int d  = n & 63;
        ((__bf16*)out)[(((size_t)(b_ * HEADS + hh)) * HDIM + d) * SEQ + s_] =
            (__bf16)v;
      } else {
        ((float*)out)[(size_t)m * N + n] = resid[(size_t)m * N + n] + v;
      }
    }
  }
}

// ---------------------------------------------------------------------------
// Flash-style causal attention. One wave (block=32) per 16-query tile.
// Q,K: bf16 [NTOK, EMB] (head cols contiguous). Vt: bf16 [B][H][64][SEQ].
// ctx out: bf16 [NTOK, EMB]. __launch_bounds__(32) -> full VGPR budget,
// keeps Q fragments and O accumulators resident (no scratch spills).
// ---------------------------------------------------------------------------
__global__ __launch_bounds__(32) void k_attn(const __bf16* __restrict__ Q,
                                             const __bf16* __restrict__ Kb,
                                             const __bf16* __restrict__ Vt,
                                             __bf16* __restrict__ ctx) {
  const int qt = blockIdx.x;            // query tile
  const int bh = blockIdx.y;            // batch*head
  const int b  = bh >> 4;
  const int h  = bh & 15;
  const int lane = threadIdx.x & 31;
  const int hl = lane >> 4;
  const int l16 = lane & 15;
  const int q0 = qt * 16;

  __shared__ float sP[16][33];          // softmax probs, padded

  // Q fragments for K-dim (head dim) 0..31 and 32..63
  const __bf16* qrow = Q + ((size_t)(b * SEQ) + q0 + l16) * EMB + h * HDIM;
  v8bf q00 = *(const v8bf*)(qrow + hl * 16);
  v8bf q01 = *(const v8bf*)(qrow + hl * 16 + 8);
  v16bf qa0 = __builtin_shufflevector(q00, q01, 0, 1, 2, 3, 4, 5, 6, 7,
                                      8, 9, 10, 11, 12, 13, 14, 15);
  v8bf q10 = *(const v8bf*)(qrow + 32 + hl * 16);
  v8bf q11 = *(const v8bf*)(qrow + 32 + hl * 16 + 8);
  v16bf qa1 = __builtin_shufflevector(q10, q11, 0, 1, 2, 3, 4, 5, 6, 7,
                                      8, 9, 10, 11, 12, 13, 14, 15);

  float mrun[8], lrun[8];
  v8f o0 = {}, o1 = {}, o2 = {}, o3 = {};
#pragma unroll
  for (int r = 0; r < 8; r++) { mrun[r] = -3e38f; lrun[r] = 0.f; }

  const float scale = 0.125f;           // 1/sqrt(64)

  for (int kb = 0; kb < q0 + 16; kb += 32) {
    // ---- scores: two 16x16 tiles over key block kb..kb+31 ----
    v8f sc0, sc1;
    {
      const __bf16* krow0 =
          Kb + ((size_t)(b * SEQ) + kb + l16) * EMB + h * HDIM;
      const __bf16* krow1 = krow0 + (size_t)16 * EMB;
      v16bf k00 = *(const v16bf*)(krow0 + hl * 16);
      v16bf k01 = *(const v16bf*)(krow0 + 32 + hl * 16);
      v16bf k10 = *(const v16bf*)(krow1 + hl * 16);
      v16bf k11 = *(const v16bf*)(krow1 + 32 + hl * 16);
      v8f t0 = {}, t1 = {};
      t0 = wmma_bf16(qa0, k00, t0);
      t0 = wmma_bf16(qa1, k01, t0);
      t1 = wmma_bf16(qa0, k10, t1);
      t1 = wmma_bf16(qa1, k11, t1);
      sc0 = t0; sc1 = t1;
    }
    // ---- online softmax update (row-wise over 16-lane halves) ----
#pragma unroll
    for (int r = 0; r < 8; r++) {
      const int qrow_i = q0 + r + hl * 8;
      float s0 = (kb + l16 <= qrow_i) ? sc0[r] * scale : -3e38f;
      float s1 = (kb + 16 + l16 <= qrow_i) ? sc1[r] * scale : -3e38f;
      float mx = fmaxf(s0, s1);
#pragma unroll
      for (int off = 8; off >= 1; off >>= 1)
        mx = fmaxf(mx, __shfl_xor(mx, off, 32));
      const float mnew = fmaxf(mrun[r], mx);
      const float corr = __expf(mrun[r] - mnew);
      const float p0 = __expf(s0 - mnew);
      const float p1 = __expf(s1 - mnew);
      float rowsum = p0 + p1;
#pragma unroll
      for (int off = 8; off >= 1; off >>= 1)
        rowsum += __shfl_xor(rowsum, off, 32);
      lrun[r] = lrun[r] * corr + rowsum;
      mrun[r] = mnew;
      o0[r] *= corr; o1[r] *= corr; o2[r] *= corr; o3[r] *= corr;
      sP[r + hl * 8][l16] = p0;
      sP[r + hl * 8][16 + l16] = p1;
    }
    __syncthreads();
    // ---- re-fragment P (16x32) into WMMA A layout ----
    v16bf pa;
#pragma unroll
    for (int e = 0; e < 8; e++) {
      pa[e]     = (__bf16)sP[l16][hl * 16 + e];
      pa[8 + e] = (__bf16)sP[l16][hl * 16 + 8 + e];
    }
    // ---- O += P * V over 4 d-tiles ----
    const __bf16* vrow =
        Vt + (((size_t)(b * HEADS + h)) * HDIM + l16) * SEQ + kb + hl * 16;
    const size_t vstep = (size_t)16 * SEQ;
    v16bf vb0 = *(const v16bf*)vrow;
    v16bf vb1 = *(const v16bf*)(vrow + vstep);
    v16bf vb2 = *(const v16bf*)(vrow + 2 * vstep);
    v16bf vb3 = *(const v16bf*)(vrow + 3 * vstep);
    o0 = wmma_bf16(pa, vb0, o0);
    o1 = wmma_bf16(pa, vb1, o1);
    o2 = wmma_bf16(pa, vb2, o2);
    o3 = wmma_bf16(pa, vb3, o3);
    __syncthreads();
  }

  // ---- normalize + write ctx ----
  v8f oo[4] = {o0, o1, o2, o3};
#pragma unroll
  for (int t = 0; t < 4; t++) {
#pragma unroll
    for (int r = 0; r < 8; r++) {
      const int q = q0 + r + hl * 8;
      const float val = oo[t][r] / lrun[r];
      ctx[((size_t)(b * SEQ) + q) * EMB + h * HDIM + t * 16 + l16] =
          (__bf16)val;
    }
  }
}

// ---------------------------------------------------------------------------
// Launcher
// ---------------------------------------------------------------------------
extern "C" void kernel_launch(void* const* d_in, const int* in_sizes, int n_in,
                              void* d_out, int out_size, void* d_ws,
                              size_t ws_size, hipStream_t stream) {
  const float* x  = (const float*)d_in[0];
  const float* Wq = (const float*)d_in[1];
  const float* Wk = (const float*)d_in[2];
  const float* Wv = (const float*)d_in[3];
  const float* Wo = (const float*)d_in[4];
  const float* bo = (const float*)d_in[5];
  const float* W1 = (const float*)d_in[6];
  const float* b1 = (const float*)d_in[7];
  const float* W2 = (const float*)d_in[8];
  const float* b2 = (const float*)d_in[9];
  const float* g1 = (const float*)d_in[10];
  const float* s1 = (const float*)d_in[11];
  const float* g2 = (const float*)d_in[12];
  const float* s2 = (const float*)d_in[13];
  float* outp = (float*)d_out;

  char* ws = (char*)d_ws;
  size_t off = 0;
  auto alloc = [&](size_t bytes) -> char* {
    char* p = ws + off;
    off += (bytes + 255) & ~(size_t)255;
    return p;
  };
  __bf16* WqT = (__bf16*)alloc((size_t)EMB * EMB * 2);
  __bf16* WkT = (__bf16*)alloc((size_t)EMB * EMB * 2);
  __bf16* WvT = (__bf16*)alloc((size_t)EMB * EMB * 2);
  __bf16* WoT = (__bf16*)alloc((size_t)EMB * EMB * 2);
  __bf16* W1T = (__bf16*)alloc((size_t)EMB * DFF * 2);
  __bf16* W2T = (__bf16*)alloc((size_t)DFF * EMB * 2);
  __bf16* hbf = (__bf16*)alloc((size_t)NTOK * EMB * 2);
  __bf16* Qb  = (__bf16*)alloc((size_t)NTOK * EMB * 2);
  __bf16* Kbf = (__bf16*)alloc((size_t)NTOK * EMB * 2);
  __bf16* Vt  = (__bf16*)alloc((size_t)NTOK * EMB * 2);
  __bf16* ctx = (__bf16*)alloc((size_t)NTOK * EMB * 2);
  float*  x1  = (float*)alloc((size_t)NTOK * EMB * 4);
  __bf16* h2  = (__bf16*)alloc((size_t)NTOK * EMB * 2);
  __bf16* f1  = (__bf16*)alloc((size_t)NTOK * DFF * 2);

  const int nEE = EMB * EMB;
  const int nEF = EMB * DFF;
  k_cvt_transpose<<<(nEE + 255) / 256, 256, 0, stream>>>(Wq, WqT, EMB, EMB);
  k_cvt_transpose<<<(nEE + 255) / 256, 256, 0, stream>>>(Wk, WkT, EMB, EMB);
  k_cvt_transpose<<<(nEE + 255) / 256, 256, 0, stream>>>(Wv, WvT, EMB, EMB);
  k_cvt_transpose<<<(nEE + 255) / 256, 256, 0, stream>>>(Wo, WoT, EMB, EMB);
  k_cvt_transpose<<<(nEF + 255) / 256, 256, 0, stream>>>(W1, W1T, EMB, DFF);
  k_cvt_transpose<<<(nEF + 255) / 256, 256, 0, stream>>>(W2, W2T, DFF, EMB);

  // LN1
  k_layernorm<<<NTOK, 256, 0, stream>>>(x, g1, s1, hbf);

  // Q, K, V projections  (block tile 32 x 256)
  dim3 gEE(EMB / 256, NTOK / 32);
  k_gemm<0, false><<<gEE, 256, 0, stream>>>(hbf, WqT, nullptr, nullptr, Qb,
                                            NTOK, EMB, EMB);
  k_gemm<0, false><<<gEE, 256, 0, stream>>>(hbf, WkT, nullptr, nullptr, Kbf,
                                            NTOK, EMB, EMB);
  k_gemm<1, false><<<gEE, 256, 0, stream>>>(hbf, WvT, nullptr, nullptr, Vt,
                                            NTOK, EMB, EMB);

  // attention
  dim3 gA(SEQ / 16, BATCH * HEADS);
  k_attn<<<gA, 32, 0, stream>>>(Qb, Kbf, Vt, ctx);

  // output projection + residual: x1 = x + ctx @ Wo + bo
  k_gemm<2, false><<<gEE, 256, 0, stream>>>(ctx, WoT, bo, x, x1, NTOK, EMB,
                                            EMB);

  // LN2
  k_layernorm<<<NTOK, 256, 0, stream>>>(x1, g2, s2, h2);

  // FFN1: gelu(h2 @ W1 + b1)
  dim3 gEF(DFF / 256, NTOK / 32);
  k_gemm<0, true><<<gEF, 256, 0, stream>>>(h2, W1T, b1, nullptr, f1, NTOK,
                                           DFF, EMB);
  // FFN2: out = x1 + f1 @ W2 + b2
  k_gemm<2, false><<<gEE, 256, 0, stream>>>(f1, W2T, b2, x1, outp, NTOK, EMB,
                                            DFF);
}
